// GAT_9096740733072
// MI455X (gfx1250) — matrix operand order repaired and checked by reading, not compiled, not run on hardware
//
#include <hip/hip_runtime.h>

// ---------------------------------------------------------------------------
// Fused 2-layer GAT for MI455X (gfx1250, wave32, WMMA bf16 16x16x32).
// One workgroup (8 waves) per batch element; re=[r|e] staged once in LDS as
// bf16 and reused by both layers' GEMMs; per-b constant blocks of the concat
// are folded into the accumulator bias. Compute-bound -> bf16 WMMA path.
// ---------------------------------------------------------------------------

typedef __attribute__((ext_vector_type(16))) __bf16 v16bf;
typedef __attribute__((ext_vector_type(8)))  __bf16 v8bf;
typedef __attribute__((ext_vector_type(4)))  __bf16 v4bf;
typedef __attribute__((ext_vector_type(8)))  float  v8f;
typedef __attribute__((ext_vector_type(4)))  float  v4f;

#define B_    2048
#define N_    256
#define E_    128
#define R_    128
#define H_    4
#define D_    128
#define F0_   384     // 2E+R   (ff0_W row length)
#define F1_   768     // H*D+E+R (ff1_W row length)
#define KRE   256     // K of the real GEMM ([r|e] width)
#define RE_STR 264    // padded LDS row stride (bf16) for re  -> conflict-free b128 loads
#define X_STR  132    // padded LDS row stride (f32) for x

// LDS layout (bytes)
#define RE_BYTES   (N_ * RE_STR * 2)          // 135168
#define X_BYTES    (N_ * X_STR * 4)           // 135168
#define OFF_HVEC   (RE_BYTES + X_BYTES)       // 270336
#define OFF_H2     (OFF_HVEC + 128 * 4)
#define OFF_YV     (OFF_H2   + 512 * 4)
#define OFF_ATTN   (OFF_YV   + 128 * 4)
#define OFF_RED    (OFF_ATTN + 1024 * 4)
#define OFF_AGG    (OFF_RED  + 128 * 4)
#define SMEM_BYTES (OFF_AGG  + 512 * 4)       // 280064 < 320KB

static __device__ __forceinline__ __bf16 f2bf(float f) {
  unsigned u = __builtin_bit_cast(unsigned, f);
  unsigned r = (u + 0x7FFFu + ((u >> 16) & 1u)) >> 16;   // round-to-nearest-even
  unsigned short hs = (unsigned short)r;
  return __builtin_bit_cast(__bf16, hs);
}

// GEMM: x(256x128,f32,LDS) = re(256x256,bf16,LDS) @ W[:, f0:f0+256]^T + yv[d]
// Wave w owns output columns d in [16w, 16w+16). K = 256 -> 8 WMMA steps.
static __device__ __forceinline__ void layer_gemm(
    const __bf16* re, float* x, const float* yv,
    const float* W, int f0, int Fld, int lane, int wave)
{
  const int dcol = wave * 16 + (lane & 15);
  // B-matrix (32x16 bf16): lane (lane&15) = output column N; each half-wave
  // holds a 16-long K run (K = (lane>>4)*16 + i), i.e. 16 contiguous elements
  // of W row 'dcol' -> load fp32, convert, keep in registers for all n-tiles.
  const int kk = (lane >> 4) * 16;
  v16bf bfrag[8];
  const float* wb = W + (size_t)dcol * Fld + f0 + kk;
#pragma unroll
  for (int s = 0; s < 8; ++s) {
    v16bf bf;
#pragma unroll
    for (int i = 0; i < 16; ++i) bf[i] = f2bf(wb[s * 32 + i]);
    bfrag[s] = bf;
  }
  const float bias = yv[dcol];
  const int mlo   = lane & 15;        // A row within tile
  const int kbase = (lane >> 4) * 16; // A-layout: half-wave contiguous 16-K run
  for (int nt = 0; nt < 16; ++nt) {
    v8f acc = {bias, bias, bias, bias, bias, bias, bias, bias};
    const __bf16* arow = re + (size_t)(nt * 16 + mlo) * RE_STR + kbase;
#pragma unroll
    for (int s = 0; s < 8; ++s) {
      // A (16x32 bf16): elems 0..15 = K kbase..kbase+15
      v8bf lo = *(const v8bf*)(arow + s * 32);
      v8bf hi = *(const v8bf*)(arow + s * 32 + 8);
      v16bf a = __builtin_shufflevector(lo, hi, 0, 1, 2, 3, 4, 5, 6, 7,
                                        8, 9, 10, 11, 12, 13, 14, 15);
      acc = __builtin_amdgcn_wmma_f32_16x16x32_bf16(
          false, a, false, bfrag[s], (short)0, acc, false, false);
    }
    // D tile (16x16 f32): lanes 0-15 -> M=j, lanes 16-31 -> M=8+j; N=lane&15
    const int mhi = (lane >> 4) * 8;
#pragma unroll
    for (int j = 0; j < 8; ++j)
      x[(size_t)(nt * 16 + mhi + j) * X_STR + dcol] = acc[j];
  }
}

// logits -> leaky(0.1) -> mask -> softmax over n; result into attnbuf[n*4+h]
static __device__ __forceinline__ void attention(
    const float* x, float* attnbuf, float* redbuf,
    const float* aW, const float* ab, const float* masks_row, int t)
{
  {
    const int n = t;
    const float mterm = 1e31f * (1.0f - masks_row[n]);
    const float* xr = x + (size_t)n * X_STR;
#pragma unroll
    for (int h = 0; h < H_; ++h) {
      float acc = ab[h];
      const float* ar = aW + h * D_;
      for (int d = 0; d < D_; ++d) acc += xr[d] * ar[d];
      acc = acc >= 0.0f ? acc : 0.1f * acc;
      attnbuf[n * 4 + h] = acc - mterm;
    }
  }
  __syncthreads();
  if (t < 64) {                       // partial max: h = t>>4, 16 n's each
    const int h = t >> 4, seg = t & 15;
    float m = -3.4e38f;
    for (int i = 0; i < 16; ++i) m = fmaxf(m, attnbuf[(seg * 16 + i) * 4 + h]);
    redbuf[t] = m;
  }
  __syncthreads();
  if (t < 4) {
    float m = -3.4e38f;
    for (int i = 0; i < 16; ++i) m = fmaxf(m, redbuf[t * 16 + i]);
    redbuf[64 + t] = m;
  }
  __syncthreads();
  if (t < 64) {                       // partial sum of exp
    const int h = t >> 4, seg = t & 15;
    const float gm = redbuf[64 + h];
    float s = 0.0f;
    for (int i = 0; i < 16; ++i) s += __expf(attnbuf[(seg * 16 + i) * 4 + h] - gm);
    redbuf[t] = s;
  }
  __syncthreads();
  if (t < 4) {
    float s = 0.0f;
    for (int i = 0; i < 16; ++i) s += redbuf[t * 16 + i];
    redbuf[68 + t] = s;
  }
  __syncthreads();
  {
    const int n = t;
#pragma unroll
    for (int h = 0; h < H_; ++h) {
      const float gm = redbuf[64 + h], gs = redbuf[68 + h];
      attnbuf[n * 4 + h] = __expf(attnbuf[n * 4 + h] - gm) / gs;
    }
  }
  __syncthreads();
}

// agg[h][d] = sum_n attn[n][h] * x[n][d]   (each thread: 2 (h,d) outputs)
static __device__ __forceinline__ void aggregate(
    const float* x, const float* attnbuf, float* aggs, int t)
{
  const int d = t & 127;
  const int hb = t >> 7;
#pragma unroll
  for (int p = 0; p < 2; ++p) {
    const int h = hb + 2 * p;
    float s = 0.0f;
    for (int n = 0; n < N_; ++n)
      s += attnbuf[n * 4 + h] * x[(size_t)n * X_STR + d];
    aggs[h * 128 + d] = s;
  }
  __syncthreads();
}

__global__ void __launch_bounds__(256) gat_fused(
    const int* __restrict__ batch_e1, const int* __restrict__ neighbors,
    const float* __restrict__ masks,
    const float* __restrict__ emb_e, const float* __restrict__ emb_r,
    const float* __restrict__ ff0_W, const float* __restrict__ ff0_b,
    const float* __restrict__ ff1_W, const float* __restrict__ ff1_b,
    const float* __restrict__ a0_W, const float* __restrict__ a0_b,
    const float* __restrict__ a1_W, const float* __restrict__ a1_b,
    float* __restrict__ out)
{
  const int b    = blockIdx.x;
  const int t    = threadIdx.x;   // 0..255
  const int lane = t & 31;
  const int wave = t >> 5;        // 0..7

  extern __shared__ char smem[];
  __bf16* re      = (__bf16*)smem;
  float*  x       = (float*)(smem + RE_BYTES);
  float*  hvec    = (float*)(smem + OFF_HVEC);
  float*  h2      = (float*)(smem + OFF_H2);
  float*  yv      = (float*)(smem + OFF_YV);
  float*  attnbuf = (float*)(smem + OFF_ATTN);
  float*  redbuf  = (float*)(smem + OFF_RED);
  float*  aggs    = (float*)(smem + OFF_AGG);

  // ---- stage h and re = [r_n | e_n] as bf16 ------------------------------
  if (t < 128) hvec[t] = emb_e[(size_t)batch_e1[b] * E_ + t];
  for (int i = 0; i < 32; ++i) {        // wave handles 32 rows, 1 float4/lane
    const int n  = wave * 32 + i;
    const int ir = neighbors[((size_t)b * N_ + n) * 2 + 0];
    const int ie = neighbors[((size_t)b * N_ + n) * 2 + 1];
    v4f rr = *(const v4f*)(emb_r + (size_t)ir * R_ + lane * 4);
    v4f ee = *(const v4f*)(emb_e + (size_t)ie * E_ + lane * 4);
    __bf16* dst = re + (size_t)n * RE_STR;
    v4bf rb = {f2bf(rr.x), f2bf(rr.y), f2bf(rr.z), f2bf(rr.w)};
    v4bf eb = {f2bf(ee.x), f2bf(ee.y), f2bf(ee.z), f2bf(ee.w)};
    *(v4bf*)(dst + lane * 4)        = rb;
    *(v4bf*)(dst + 128 + lane * 4)  = eb;
  }
  __syncthreads();

  // ---- layer 1: y0[d] = ff0_b[d] + h . ff0_W[d, 0:128] -------------------
  if (t < 128) {
    float acc = ff0_b[t];
    const float* wr = ff0_W + (size_t)t * F0_;
    for (int f = 0; f < 128; ++f) acc += hvec[f] * wr[f];
    yv[t] = acc;
  }
  __syncthreads();

  layer_gemm(re, x, yv, ff0_W, /*f0=*/128, F0_, lane, wave);   // x1
  __syncthreads();

  attention(x, attnbuf, redbuf, a0_W, a0_b, masks + (size_t)b * N_, t);
  aggregate(x, attnbuf, aggs, t);

  // h2 = leaky(agg1, 0.01), flat 512
#pragma unroll
  for (int p = 0; p < 2; ++p) {
    const int j = t + p * 256;
    const float v = aggs[j];
    h2[j] = v >= 0.0f ? v : 0.01f * v;
  }
  __syncthreads();

  // ---- layer 2: y1[d] = ff1_b[d] + h2 . ff1_W[d, 0:512] ------------------
  if (t < 128) {
    float acc = ff1_b[t];
    const float* wr = ff1_W + (size_t)t * F1_;
    for (int f = 0; f < 512; ++f) acc += h2[f] * wr[f];
    yv[t] = acc;
  }
  __syncthreads();

  layer_gemm(re, x, yv, ff1_W, /*f0=*/512, F1_, lane, wave);   // x2 (reuses x)
  __syncthreads();

  attention(x, attnbuf, redbuf, a1_W, a1_b, masks + (size_t)b * N_, t);
  aggregate(x, attnbuf, aggs, t);

  // ---- out[b][d] = h[d] + mean_h agg2[h][d] ------------------------------
  if (t < 128) {
    const float s = aggs[t] + aggs[128 + t] + aggs[256 + t] + aggs[384 + t];
    out[(size_t)b * D_ + t] = hvec[t] + 0.25f * s;
  }
}

extern "C" void kernel_launch(void* const* d_in, const int* in_sizes, int n_in,
                              void* d_out, int out_size, void* d_ws, size_t ws_size,
                              hipStream_t stream) {
  (void)in_sizes; (void)n_in; (void)out_size; (void)d_ws; (void)ws_size;
  const int*   batch_e1  = (const int*)d_in[0];
  const int*   batch_q   = (const int*)d_in[1];  (void)batch_q; // unused by ref
  const int*   neighbors = (const int*)d_in[2];
  const float* masks     = (const float*)d_in[3];
  const float* emb_e     = (const float*)d_in[4];
  const float* emb_r     = (const float*)d_in[5];
  const float* ff0_W     = (const float*)d_in[6];
  const float* ff0_b     = (const float*)d_in[7];
  const float* ff1_W     = (const float*)d_in[8];
  const float* ff1_b     = (const float*)d_in[9];
  const float* a0_W      = (const float*)d_in[10];
  const float* a0_b      = (const float*)d_in[11];
  const float* a1_W      = (const float*)d_in[12];
  const float* a1_b      = (const float*)d_in[13];
  float* out = (float*)d_out;

  hipFuncSetAttribute((const void*)gat_fused,
                      hipFuncAttributeMaxDynamicSharedMemorySize, SMEM_BYTES);
  gat_fused<<<B_, 256, SMEM_BYTES, stream>>>(
      batch_e1, neighbors, masks, emb_e, emb_r,
      ff0_W, ff0_b, ff1_W, ff1_b, a0_W, a0_b, a1_W, a1_b, out);
}
